// MultipleSequenceAlignment_72249939853676
// MI455X (gfx1250) — compile-verified, matrix operand order checked by reading, not run on hardware
//
#include <hip/hip_runtime.h>

// ---------------------------------------------------------------------------
// AF3 MSA module block on gfx1250 (MI455X). All heavy einsums lower to one
// generalized-stride WMMA GEMM kernel (v_wmma_f32_16x16x32_f16, f32 accum).
// ~270 GFLOP; working set fits in 192MB L2 -> compute bound -> f16 matrix
// pipe.  Addressing uses shift/mask dim decomposition (no integer division);
// tile staging uses pipelined 16-byte vector loads whenever the descriptor
// strides allow (covers ~95% of the FLOPs), guarded scalar gathers otherwise.
// Workspace use ~290 MB.
// ---------------------------------------------------------------------------

typedef _Float16 half_t;
typedef __attribute__((ext_vector_type(16))) _Float16 v16h;
typedef __attribute__((ext_vector_type(8)))  _Float16 v8h;
typedef __attribute__((ext_vector_type(8)))  float    v8f;

#define S_RES 256
#define NSEQ  512
#define CM    64
#define CZ    128
#define RM    (S_RES * NSEQ)                // 131072 MSA rows
#define RZ    (S_RES * S_RES)               // 65536 pair rows
static const long long NZ = (long long)RZ * CZ;   // 8388608
static const long long NM = (long long)RM * CM;   // 8388608

// addr(v) = (v>>sh)*s1 + (v&mask)*s0 -- one GEMM kernel expresses every
// einsum layout in this model (fused dims, transposes, per-head batching).
// All "div" factors in this model are powers of two -> pure shift/mask.
struct Dim2 { long long s1, s0, mask; int sh; };

struct GemmDesc {
  const half_t* A; const half_t* B;
  float* C; half_t* Ch;                     // exactly one of C/Ch non-null
  int M, N, K;
  Dim2 aM, aK, bK, bN, cM, cN, batA, batB, batC;
  float alpha; int accum;
  int fA, fB;                               // vectorized staging fast paths
};

__device__ __forceinline__ long long ixd(long long v, Dim2 d) {
  return (v >> d.sh) * d.s1 + (v & d.mask) * d.s0;
}

#define BM 128
#define BN 64
#define BK 32
#define PAD 8                               // keep LDS rows 16B aligned

__global__ __launch_bounds__(128) void k_gemm(GemmDesc g) {
  __shared__ half_t sA[BM][BK + PAD];
  __shared__ half_t sB[BN][BK + PAD];     // B stored transposed: sB[n][k]
  const int tid  = threadIdx.x;
  const int lane = tid & 31;
  const int wave = tid >> 5;               // 4 waves, each owns a 32x64 strip
  const long long bz = blockIdx.z;
  const long long offA = ixd(bz, g.batA);
  const long long offB = ixd(bz, g.batB);
  const long long offC = ixd(bz, g.batC);
  const int m0 = blockIdx.y * BM;
  const int n0 = blockIdx.x * BN;

  // fixed per-thread staging coordinates (scalar paths)
  const int cA  = tid & 31;   // A: k-column within tile
  const int rA0 = tid >> 5;   // A: row step base (t*4 + rA0)
  const int cB  = tid & 63;   // B: n-column within tile
  const int rB0 = tid >> 6;   // B: k step base (t*2 + rB0)
  const bool nBok = (n0 + cB) < g.N;
  const long long bColOff = nBok ? ixd(n0 + cB, g.bN) : 0;   // loop-invariant

  v8f acc[2][4] = {};

  for (int k0 = 0; k0 < g.K; k0 += BK) {
    if (g.fA) {
      // vector path: 128 rows x 4 x v8h; thread -> (row = v>>2, kgroup = v&3)
      // load-all-then-store-all so the 4 global_load_b128 pipeline.
      v8h ra[4];
#pragma unroll
      for (int t = 0; t < 4; ++t) {
        int v = t * 128 + tid;
        int r = v >> 2, kg = v & 3;
        long long ao = offA + ixd(m0 + r, g.aM) + ixd(k0 + kg * 8, g.aK);
        ra[t] = *(const v8h*)(g.A + ao);
        if (k0 + BK < g.K) __builtin_prefetch((const void*)(g.A + ao + BK), 0, 1);
      }
#pragma unroll
      for (int t = 0; t < 4; ++t) {
        int v = t * 128 + tid;
        int r = v >> 2, kg = v & 3;
        *(v8h*)&sA[r][kg * 8] = ra[t];
      }
    } else {
      const int kA = k0 + cA;
      const bool kAok = kA < g.K;
      const long long aColOff = kAok ? ixd(kA, g.aK) : 0;
#pragma unroll
      for (int t = 0; t < 32; ++t) {
        int r = t * 4 + rA0;
        int m = m0 + r;
        half_t v = (half_t)0.f;
        if (m < g.M && kAok) v = g.A[offA + ixd(m, g.aM) + aColOff];
        sA[r][cA] = v;
      }
    }
    if (g.fB) {
      // vector path along n: 32 k-rows x 8 x v8h; register transpose to sB[n][k]
      v8h rb[2];
#pragma unroll
      for (int t = 0; t < 2; ++t) {
        int v = t * 128 + tid;
        int kk = v >> 3, ng = v & 7;
        long long bo = offB + ixd(k0 + kk, g.bK) + ixd(n0 + ng * 8, g.bN);
        rb[t] = *(const v8h*)(g.B + bo);
      }
#pragma unroll
      for (int t = 0; t < 2; ++t) {
        int v = t * 128 + tid;
        int kk = v >> 3, ng = v & 7;
#pragma unroll
        for (int e = 0; e < 8; ++e) sB[ng * 8 + e][kk] = rb[t][e];
      }
    } else {
#pragma unroll
      for (int t = 0; t < 16; ++t) {
        int kk = t * 2 + rB0;
        int k = k0 + kk;
        half_t v = (half_t)0.f;
        if (k < g.K && nBok) v = g.B[offB + ixd(k, g.bK) + bColOff];
        sB[cB][kk] = v;
      }
    }
    __syncthreads();

    // ---- fragments per ISA 7.12.2 wave32 layouts ----
    const int half32 = lane >> 4;
    const int rl = lane & 15;
    v16h fa[2], fb[4];
#pragma unroll
    for (int x = 0; x < 4; ++x) {
      int col = x * 16 + rl;
#pragma unroll
      for (int j = 0; j < 8; ++j) {
        int kk = ((j >> 2) << 4) + (half32 << 3) + ((j & 3) << 1);
        fb[x][2 * j]     = sB[col][kk];
        fb[x][2 * j + 1] = sB[col][kk + 1];
      }
    }
#pragma unroll
    for (int t = 0; t < 2; ++t) {
      int row = wave * 32 + t * 16 + rl;
#pragma unroll
      for (int j = 0; j < 8; ++j) {
        int kk = ((j >> 2) << 4) + (half32 << 3) + ((j & 3) << 1);
        fa[t][2 * j]     = sA[row][kk];
        fa[t][2 * j + 1] = sA[row][kk + 1];
      }
    }
#pragma unroll
    for (int ty = 0; ty < 2; ++ty)
#pragma unroll
      for (int tx = 0; tx < 4; ++tx)
        acc[ty][tx] = __builtin_amdgcn_wmma_f32_16x16x32_f16(
            false, fa[ty], false, fb[tx], (short)0, acc[ty][tx], false, false);
    __syncthreads();
  }

  // epilogue: C layout lane L vgpr r -> m = r + 8*(L>>4), n = L&15
  const int half32 = lane >> 4;
  const int nl = lane & 15;
#pragma unroll
  for (int ty = 0; ty < 2; ++ty)
#pragma unroll
    for (int tx = 0; tx < 4; ++tx)
#pragma unroll
      for (int r = 0; r < 8; ++r) {
        int m = m0 + wave * 32 + ty * 16 + r + half32 * 8;
        int n = n0 + tx * 16 + nl;
        if (m < g.M && n < g.N) {
          float v = g.alpha * acc[ty][tx][r];
          long long co = offC + ixd(m, g.cM) + ixd(n, g.cN);
          if (g.Ch) {
            g.Ch[co] = (half_t)v;
          } else {
            if (g.accum) v += g.C[co];
            g.C[co] = v;
          }
        }
      }
}

// ---------------------------------------------------------------------------
// support kernels (VALU work: LN, softmax, gating, packing)
// ---------------------------------------------------------------------------

__global__ void k_cvt(const float* __restrict__ s, half_t* __restrict__ d, long long n) {
  long long i = (long long)blockIdx.x * 256 + threadIdx.x;
  if (i < n) d[i] = (half_t)s[i];
}

__global__ void k_pack25(const float* __restrict__ msa, const float* __restrict__ hd,
                         const float* __restrict__ dv, half_t* __restrict__ o, long long rows) {
  long long r = (long long)blockIdx.x * 256 + threadIdx.x;
  if (r < rows) {
    for (int c = 0; c < 23; ++c) o[r * 25 + c] = (half_t)msa[r * 23 + c];
    o[r * 25 + 23] = (half_t)hd[r];
    o[r * 25 + 24] = (half_t)dv[r];
  }
}

// blockDim.x == C (64 or 128); one row per block
__global__ void k_ln(const float* __restrict__ x, const float* __restrict__ g,
                     const float* __restrict__ b, half_t* __restrict__ y, int C) {
  __shared__ float red[128];
  long long row = blockIdx.x;
  int c = threadIdx.x;
  float v = x[row * C + c];
  red[c] = v; __syncthreads();
  for (int s = blockDim.x / 2; s > 0; s >>= 1) { if (c < s) red[c] += red[c + s]; __syncthreads(); }
  float mu = red[0] / C; __syncthreads();
  float d = v - mu;
  red[c] = d * d; __syncthreads();
  for (int s = blockDim.x / 2; s > 0; s >>= 1) { if (c < s) red[c] += red[c + s]; __syncthreads(); }
  float var = red[0] / C;
  y[row * C + c] = (half_t)(d * rsqrtf(var + 1e-5f) * g[c] + b[c]);
}

// softmax over j of bias (i,j,h) f32 -> f16 ; grid = S*8, block = 256 (j)
__global__ void k_softmax_mpwa(const float* __restrict__ bias, half_t* __restrict__ w) {
  __shared__ float red[256];
  int i = blockIdx.x >> 3, h = blockIdx.x & 7, j = threadIdx.x;
  long long idx = ((long long)(i * 256 + j)) * 8 + h;
  float v = bias[idx];
  red[j] = v; __syncthreads();
  for (int s = 128; s > 0; s >>= 1) { if (j < s) red[j] = fmaxf(red[j], red[j + s]); __syncthreads(); }
  float mx = red[0]; __syncthreads();
  float e = expf(v - mx);
  red[j] = e; __syncthreads();
  for (int s = 128; s > 0; s >>= 1) { if (j < s) red[j] += red[j + s]; __syncthreads(); }
  w[idx] = (half_t)(e / red[0]);
}

// attention softmax over k with bias bb[(j*256+k)*4+h]; grid = 128*256 rows
__global__ void k_softmax_attn(const float* __restrict__ lg, const float* __restrict__ bb,
                               half_t* __restrict__ w) {
  __shared__ float red[256];
  int row = blockIdx.x;              // bz*256 + j, bz = i_loc*4 + h
  int bz = row >> 8, j = row & 255, h = bz & 3, k = threadIdx.x;
  long long off = (long long)row * 256 + k;
  float v = lg[off] + bb[((long long)(j * 256 + k)) * 4 + h];
  red[k] = v; __syncthreads();
  for (int s = 128; s > 0; s >>= 1) { if (k < s) red[k] = fmaxf(red[k], red[k + s]); __syncthreads(); }
  float mx = red[0]; __syncthreads();
  float e = expf(v - mx);
  red[k] = e; __syncthreads();
  for (int s = 128; s > 0; s >>= 1) { if (k < s) red[k] += red[k + s]; __syncthreads(); }
  w[off] = (half_t)(e / red[0]);
}

__global__ void k_sig_mul_f16(const float* __restrict__ a, const float* __restrict__ o,
                              half_t* __restrict__ y, long long n) {
  long long i = (long long)blockIdx.x * 256 + threadIdx.x;
  if (i < n) y[i] = (half_t)((1.f / (1.f + expf(-a[i]))) * o[i]);
}

__global__ void k_silu_mul_f16(const float* __restrict__ a, const float* __restrict__ b,
                               half_t* __restrict__ y, long long n) {
  long long i = (long long)blockIdx.x * 256 + threadIdx.x;
  if (i < n) { float x = a[i]; y[i] = (half_t)((x / (1.f + expf(-x))) * b[i]); }
}

__global__ void k_sig_mul_add(float* __restrict__ z, const float* __restrict__ g,
                              const float* __restrict__ t, long long n) {
  long long i = (long long)blockIdx.x * 256 + threadIdx.x;
  if (i < n) z[i] += (1.f / (1.f + expf(-g[i]))) * t[i];
}

// x[i] += add[((i/C)>>sh)*C + (i%C)]  with C = 64 or 128 (power of two)
__global__ void k_add_bcast(float* __restrict__ x, const float* __restrict__ add,
                            long long n, int csh, int sh) {
  long long i = (long long)blockIdx.x * 256 + threadIdx.x;
  if (i < n) {
    long long r = i >> csh;
    long long c = i & ((1 << csh) - 1);
    x[i] += add[((r >> sh) << csh) + c];
  }
}

// d[(j,i,c)] = s[(i,j,c)] for (256,256,128) f16
__global__ void k_transpose_f16(const half_t* __restrict__ s, half_t* __restrict__ d, long long n) {
  long long gid = (long long)blockIdx.x * 256 + threadIdx.x;
  if (gid < n) {
    int c = (int)(gid & 127);
    long long t = gid >> 7;
    int jj = (int)(t & 255);
    long long ii = t >> 8;
    d[((long long)jj * 256 + ii) * 128 + c] = s[gid];
  }
}

// ---------------------------------------------------------------------------
// host side
// ---------------------------------------------------------------------------

static inline Dim2 d1(long long s) {
  Dim2 d; d.s1 = 0; d.s0 = s; d.sh = 40; d.mask = (1LL << 40) - 1; return d;
}
static inline Dim2 ds(int sh, long long s1, long long s0) {
  Dim2 d; d.s1 = s1; d.s0 = s0; d.sh = sh; d.mask = (1LL << sh) - 1; return d;
}

static void gemm_full(hipStream_t st, const half_t* A, const half_t* B, float* C, half_t* Ch,
                      int M, int N, int K, Dim2 aM, Dim2 aK, Dim2 bK, Dim2 bN, Dim2 cM, Dim2 cN,
                      int nbat, Dim2 bA, Dim2 bB, Dim2 bC, float alpha, int accum) {
  GemmDesc g;
  g.A = A; g.B = B; g.C = C; g.Ch = Ch;
  g.M = M; g.N = N; g.K = K;
  g.aM = aM; g.aK = aK; g.bK = bK; g.bN = bN; g.cM = cM; g.cN = cN;
  g.batA = bA; g.batB = bB; g.batC = bC;
  g.alpha = alpha; g.accum = accum;
  bool kFull = (K % BK) == 0;
  // A fast path: unit-stride k in 8-aligned groups, all offsets 16B-aligned.
  g.fA = (kFull && (M % BM) == 0 && aK.s0 == 1 && ((aK.mask + 1) & 7) == 0 &&
          (aK.s1 % 8) == 0 && (aM.s0 % 8) == 0 && (aM.s1 % 8) == 0 &&
          (bA.s0 % 8) == 0 && (bA.s1 % 8) == 0) ? 1 : 0;
  // B fast path: unit-stride n in 8-aligned groups, all offsets 16B-aligned.
  g.fB = (kFull && (N % BN) == 0 && bN.s0 == 1 && ((bN.mask + 1) & 7) == 0 &&
          (bN.s1 % 8) == 0 && (bK.s0 % 8) == 0 && (bK.s1 % 8) == 0 &&
          (bB.s0 % 8) == 0 && (bB.s1 % 8) == 0) ? 1 : 0;
  dim3 gr((unsigned)((N + BN - 1) / BN), (unsigned)((M + BM - 1) / BM), (unsigned)nbat);
  k_gemm<<<gr, 128, 0, st>>>(g);
}

static void gemm_rm(hipStream_t st, const half_t* A, int lda, const half_t* B, int ldb,
                    float* C, half_t* Ch, int ldc, int M, int N, int K, float alpha, int accum) {
  gemm_full(st, A, B, C, Ch, M, N, K, d1(lda), d1(1), d1(ldb), d1(1), d1(ldc), d1(1),
            1, d1(0), d1(0), d1(0), alpha, accum);
}

extern "C" void kernel_launch(void* const* d_in, const int* in_sizes, int n_in,
                              void* d_out, int out_size, void* d_ws, size_t ws_size,
                              hipStream_t stream) {
  (void)in_sizes; (void)n_in; (void)out_size; (void)ws_size;

  auto IN = [&](int i) -> const float* { return (const float*)d_in[i]; };

  const float* f_msa = IN(0);
  const float* f_hd  = IN(1);
  const float* f_dv  = IN(2);
  const float* s_in  = IN(3);
  const float* z_in  = IN(4);
  // params flattened jax-pytree style: dict keys sorted (capitals first).
  // 5 W_msa, 6 W_single,
  // mpwa: 7 Wb 8 Wg 9 Wo 10 Wv 11 b 12 bz 13 g 14 gz
  // msa_trans: 15 Wa 16 Wb 17 Wo 18 b 19 g
  // opm: 20 Wa 21 Wb 22 Wo 23 b 24 bo 25 g
  // pair_trans: 26 Wa 27 Wb 28 Wo 29 b 30 g
  // ta_end: 31 Wb2 32 Wg 33 Wk 34 Wo 35 Wq 36 Wv 37 b 38 g
  // ta_start: 39..46 (same order)
  // tm_in: 47 Wag 48 Wap 49 Wbg 50 Wbp 51 Wg 52 Wo 53 b 54 bo 55 g 56 go
  // tm_out: 57..66 (same order)

  // ---- workspace bump allocator (needs ~290 MB) ----
  char* base = (char*)d_ws;
  size_t off = 0;
  auto alloc = [&](size_t bytes) -> void* {
    void* p = base + off;
    off += (bytes + 255) & ~(size_t)255;
    return p;
  };

  float*  m     = (float*)alloc(NM * 4);
  half_t* mln   = (half_t*)alloc(NM * 2);
  half_t* zln   = (half_t*)alloc(NZ * 2);
  half_t* h1    = (half_t*)alloc(NZ * 2);
  half_t* h2    = (half_t*)alloc(NZ * 2);
  half_t* h3    = (half_t*)alloc(NZ * 2);
  half_t* h4    = (half_t*)alloc(NZ * 2);
  half_t* h5    = (half_t*)alloc(NZ * 2);
  half_t* opmA  = (half_t*)alloc((long long)RM * 32 * 2);
  half_t* opmB  = (half_t*)alloc((long long)RM * 32 * 2);
  half_t* opmC  = (half_t*)alloc((long long)512 * 8192 * 2);
  float*  buf1  = (float*)alloc(NZ * 4);
  float*  buf2  = (float*)alloc(NZ * 4);
  float*  buf3  = (float*)alloc(NZ * 4);
  half_t* in25  = (half_t*)alloc((long long)RM * 25 * 2);
  float*  bb    = (float*)alloc((long long)RZ * 4 * 4);
  half_t* wm16  = (half_t*)alloc((long long)RZ * 8 * 2);
  float*  ts    = (float*)alloc((long long)S_RES * CM * 4);

  auto cvt = [&](const float* s, long long n) -> half_t* {
    half_t* d = (half_t*)alloc(n * 2);
    k_cvt<<<(unsigned)((n + 255) / 256), 256, 0, stream>>>(s, d, n);
    return d;
  };
  auto ln = [&](const float* x, const float* g, const float* b, half_t* y, long long R, int C) {
    k_ln<<<(unsigned)R, (unsigned)C, 0, stream>>>(x, g, b, y, C);
  };
  const unsigned EB = (unsigned)(NZ / 256);   // elementwise blocks for 8.39M elems

  // ---- convert weights to f16 ----
  half_t* wmsa16    = cvt(IN(5), 25 * 64);
  half_t* wsingle16 = cvt(IN(6), 256 * 64);
  half_t* s16       = cvt(s_in, 256 * 256);
  half_t* mpWb16 = cvt(IN(7), 128 * 8);
  half_t* mpWg16 = cvt(IN(8), 64 * 64);
  half_t* mpWo16 = cvt(IN(9), 64 * 64);
  half_t* mpWv16 = cvt(IN(10), 64 * 64);
  half_t* mtWa16 = cvt(IN(15), 64 * 256);
  half_t* mtWb16 = cvt(IN(16), 64 * 256);
  half_t* mtWo16 = cvt(IN(17), 256 * 64);
  half_t* opWa16 = cvt(IN(20), 64 * 32);
  half_t* opWb16 = cvt(IN(21), 64 * 32);
  half_t* opWo16 = cvt(IN(22), 32 * 32 * 128);
  half_t* ptWa16 = cvt(IN(26), 128 * 512);
  half_t* ptWb16 = cvt(IN(27), 128 * 512);
  half_t* ptWo16 = cvt(IN(28), 512 * 128);
  half_t* taW16[2][6];   // [end=0,start=1][Wb2,Wg,Wk,Wo,Wq,Wv]
  for (int e = 0; e < 2; ++e) {
    int b0 = (e == 0) ? 31 : 39;
    taW16[e][0] = cvt(IN(b0 + 0), 128 * 4);
    for (int w = 1; w < 6; ++w) taW16[e][w] = cvt(IN(b0 + w), 128 * 128);
  }
  half_t* tmW16[2][6];   // [in=0,out=1][Wag,Wap,Wbg,Wbp,Wg,Wo]
  for (int e = 0; e < 2; ++e) {
    int b0 = (e == 0) ? 47 : 57;
    for (int w = 0; w < 6; ++w) tmW16[e][w] = cvt(IN(b0 + w), 128 * 128);
  }

  float* z = (float*)d_out;
  hipMemcpyAsync(z, z_in, NZ * 4, hipMemcpyDeviceToDevice, stream);

  // ================= m init: concat @ W_msa + broadcast(s @ W_single) =======
  k_pack25<<<(unsigned)(RM / 256), 256, 0, stream>>>(f_msa, f_hd, f_dv, in25, RM);
  gemm_rm(stream, in25, 25, wmsa16, 64, m, nullptr, 64, RM, 64, 25, 1.f, 0);
  gemm_rm(stream, s16, 256, wsingle16, 64, ts, nullptr, 64, 256, 64, 256, 1.f, 0);
  k_add_bcast<<<EB, 256, 0, stream>>>(m, ts, NM, 6, 9);        // r/512 -> sh 9

  // ================= OPM: z += outer-product-mean(m) ========================
  ln(m, IN(25), IN(23), mln, RM, 64);
  gemm_rm(stream, mln, 64, opWa16, 32, nullptr, opmA, 32, RM, 32, 64, 1.f, 0);
  gemm_rm(stream, mln, 64, opWb16, 32, nullptr, opmB, 32, RM, 32, 64, 1.f, 0);
  for (int i0 = 0; i0 < 256; i0 += 16) {
    // O[(i_loc,e),(j,f)] = (1/512) sum_n a[i,n,e] b[j,n,f]  (f16 chunk)
    gemm_full(stream, opmA + (long long)i0 * 16384, opmB, nullptr, opmC,
              512, 8192, 512,
              ds(5, 16384, 1), d1(32),
              d1(32), ds(5, 16384, 1),
              d1(8192), d1(1),
              1, d1(0), d1(0), d1(0), 1.f / 512.f, 0);
    // z[i,j,c] += sum_ef O * Wo[ef,c]
    gemm_full(stream, opmC, opWo16, z + (long long)i0 * 32768, nullptr,
              4096, 128, 1024,
              ds(8, 262144, 32), ds(5, 8192, 1),
              d1(128), d1(1),
              ds(8, 32768, 128), d1(1),
              1, d1(0), d1(0), d1(0), 1.f, 1);
  }
  k_add_bcast<<<EB, 256, 0, stream>>>(z, IN(24), NZ, 7, 40);   // + bo (r>>40 == 0)

  // ================= MPWA: m += pair-weighted averaging =====================
  ln(m, IN(13), IN(11), mln, RM, 64);
  ln(z, IN(14), IN(12), zln, RZ, 128);
  gemm_rm(stream, mln, 64, mpWv16, 64, nullptr, h3, 64, RM, 64, 64, 1.f, 0);   // v (i,n,h,d)
  gemm_rm(stream, zln, 128, mpWb16, 8, buf1, nullptr, 8, RZ, 8, 128, 1.f, 0);  // bias (i,j,h)
  gemm_rm(stream, mln, 64, mpWg16, 64, buf3, nullptr, 64, RM, 64, 64, 1.f, 0); // gate pre-sigmoid
  k_softmax_mpwa<<<2048, 256, 0, stream>>>(buf1, wm16);                         // softmax over j
  // o[i,n,h,d] = sum_j w[i,j,h] v[j,n,h,d]   (batched over h)
  gemm_full(stream, wm16, h3, buf1, nullptr, 256, 4096, 256,
            d1(2048), d1(8),
            d1(32768), ds(3, 64, 1),
            d1(32768), ds(3, 64, 1),
            8, d1(1), d1(8), d1(8), 1.f, 0);
  k_sig_mul_f16<<<EB, 256, 0, stream>>>(buf3, buf1, h1, NM);                    // g*o -> f16
  gemm_rm(stream, h1, 64, mpWo16, 64, m, nullptr, 64, RM, 64, 64, 1.f, 1);     // m += ...

  // ================= transitions (SwiGLU), row-chunked ======================
  auto transition = [&](float* x, long long R, int C, int H, const float* g, const float* b,
                        half_t* Wa, half_t* Wb, half_t* Wo, half_t* ln16) {
    ln(x, g, b, ln16, R, C);
    long long chunk = (C == 64) ? 32768 : 16384;
    for (long long r0 = 0; r0 < R; r0 += chunk) {
      gemm_rm(stream, ln16 + r0 * C, C, Wa, H, buf1, nullptr, H, (int)chunk, H, C, 1.f, 0);
      gemm_rm(stream, ln16 + r0 * C, C, Wb, H, buf2, nullptr, H, (int)chunk, H, C, 1.f, 0);
      long long n = chunk * H;
      k_silu_mul_f16<<<(unsigned)(n / 256), 256, 0, stream>>>(buf1, buf2, h1, n);
      gemm_rm(stream, h1, H, Wo, C, x + r0 * C, nullptr, C, (int)chunk, C, H, 1.f, 1);
    }
  };
  transition(m, RM, 64, 256, IN(19), IN(18), mtWa16, mtWb16, mtWo16, mln);

  // ================= triangle multiplicative updates ========================
  auto tri_mult = [&](bool outgoing, const float* g, const float* b, half_t* const* W,
                      const float* go, const float* bo) {
    ln(z, g, b, zln, RZ, 128);
    gemm_rm(stream, zln, 128, W[0], 128, buf1, nullptr, 128, RZ, 128, 128, 1.f, 0); // ag
    gemm_rm(stream, zln, 128, W[1], 128, buf2, nullptr, 128, RZ, 128, 128, 1.f, 0); // ap
    k_sig_mul_f16<<<EB, 256, 0, stream>>>(buf1, buf2, h1, NZ);
    gemm_rm(stream, zln, 128, W[2], 128, buf1, nullptr, 128, RZ, 128, 128, 1.f, 0); // bg
    gemm_rm(stream, zln, 128, W[3], 128, buf2, nullptr, 128, RZ, 128, 128, 1.f, 0); // bp
    k_sig_mul_f16<<<EB, 256, 0, stream>>>(buf1, buf2, h2, NZ);
    if (outgoing)   // s[i,j,c] = sum_k a[i,k,c] c[j,k,c]  (batched over c)
      gemm_full(stream, h1, h2, buf1, nullptr, 256, 256, 256,
                d1(32768), d1(128), d1(128), d1(32768), d1(32768), d1(128),
                128, d1(1), d1(1), d1(1), 1.f, 0);
    else            // s[i,j,c] = sum_k a[k,i,c] c[k,j,c]
      gemm_full(stream, h1, h2, buf1, nullptr, 256, 256, 256,
                d1(128), d1(32768), d1(32768), d1(128), d1(32768), d1(128),
                128, d1(1), d1(1), d1(1), 1.f, 0);
    gemm_rm(stream, zln, 128, W[4], 128, buf3, nullptr, 128, RZ, 128, 128, 1.f, 0); // gate
    ln(buf1, go, bo, h3, RZ, 128);
    gemm_rm(stream, h3, 128, W[5], 128, buf2, nullptr, 128, RZ, 128, 128, 1.f, 0);  // @Wo
    k_sig_mul_add<<<EB, 256, 0, stream>>>(z, buf3, buf2, NZ);
  };
  tri_mult(true,  IN(65), IN(63), tmW16[1], IN(66), IN(64));   // tm_out
  tri_mult(false, IN(55), IN(53), tmW16[0], IN(56), IN(54));   // tm_in

  // ================= triangle attention (start / end) =======================
  auto tri_attn = [&](bool endmode, const float* g, const float* b, half_t* const* W) {
    ln(z, g, b, zln, RZ, 128);
    half_t* zl = zln;
    if (endmode) {  // LN commutes with spatial transpose; transpose LN output
      k_transpose_f16<<<EB, 256, 0, stream>>>(zln, (half_t*)mln, NZ);
      zl = (half_t*)mln;
    }
    gemm_rm(stream, zl, 128, W[4], 128, nullptr, h1, 128, RZ, 128, 128, 1.f, 0);  // q
    gemm_rm(stream, zl, 128, W[2], 128, nullptr, h2, 128, RZ, 128, 128, 1.f, 0);  // k
    gemm_rm(stream, zl, 128, W[5], 128, nullptr, h3, 128, RZ, 128, 128, 1.f, 0);  // v
    gemm_rm(stream, zl, 128, W[1], 128, buf3, nullptr, 128, RZ, 128, 128, 1.f, 0);// gate
    gemm_rm(stream, zl, 128, W[0], 4, bb, nullptr, 4, RZ, 4, 128, 1.f, 0);        // bias
    for (int i0 = 0; i0 < 256; i0 += 32) {
      long long qo = (long long)i0 * 32768;
      // logits[bz=(i_loc,h)][j][k] = (1/sqrt(d)) q.k     (128 batches)
      gemm_full(stream, h1 + qo, h2 + qo, buf1, nullptr, 256, 256, 32,
                d1(128), d1(1),
                d1(1), d1(128),
                d1(256), d1(1),
                128, ds(2, 32768, 32), ds(2, 32768, 32), d1(65536),
                0.17677669529663687f, 0);
      k_softmax_attn<<<32768, 256, 0, stream>>>(buf1, bb, h4);
      // o[i,j,h,d] = sum_k attn * v
      gemm_full(stream, h4, h3 + qo, buf2, nullptr, 256, 32, 256,
                d1(256), d1(1),
                d1(128), d1(1),
                d1(128), d1(1),
                128, d1(65536), ds(2, 32768, 32), ds(2, 32768, 32), 1.f, 0);
      long long n = (long long)32 * 32768;
      k_sig_mul_f16<<<(unsigned)(n / 256), 256, 0, stream>>>(buf3 + qo, buf2, h5 + qo, n);
    }
    // z += (g*o) @ Wo  (end-mode: store transposed back into z)
    gemm_full(stream, h5, W[3], z, nullptr, RZ, 128, 128,
              d1(128), d1(1), d1(128), d1(1),
              endmode ? ds(8, 128, 32768) : d1(128), d1(1),
              1, d1(0), d1(0), d1(0), 1.f, 1);
  };
  tri_attn(false, IN(46), IN(45), taW16[1]);   // ta_start
  tri_attn(true,  IN(38), IN(37), taW16[0]);   // ta_end

  // ================= pair transition ========================================
  transition(z, RZ, 128, 512, IN(30), IN(29), ptWa16, ptWb16, ptWo16, zln);
}
